// SGCWGTL_77068893159665
// MI455X (gfx1250) — matrix operand order repaired and checked by reading, not compiled, not run on hardware
//
#include <hip/hip_runtime.h>

typedef __attribute__((ext_vector_type(16))) __bf16 v16bf;
typedef __attribute__((ext_vector_type(8)))  __bf16 v8bf;
typedef __attribute__((ext_vector_type(8)))  float  v8f;

#define DEV_INLINE __device__ __forceinline__

DEV_INLINE unsigned short f2bf(float x) {
  unsigned u = __float_as_uint(x);
  u += 0x7FFFu + ((u >> 16) & 1u);          // round-to-nearest-even
  return (unsigned short)(u >> 16);
}

// ---------------------------------------------------------------- utilities
__global__ void k_zero_f32(float* __restrict__ p, int n) {
  int i = blockIdx.x * blockDim.x + threadIdx.x;
  if (i < n) p[i] = 0.f;
}

__global__ void k_deg_count(const int* __restrict__ dst, float* __restrict__ deg, int E) {
  int e = blockIdx.x * blockDim.x + threadIdx.x;
  if (e < E) atomicAdd(&deg[dst[e]], 1.0f);
}

__global__ void k_dinv(float* __restrict__ deg, int n) {
  int i = blockIdx.x * blockDim.x + threadIdx.x;
  if (i < n) deg[i] = rsqrtf(deg[i] + 1.0f);    // +1 = self loop, always > 0
}

// out[i] = dinv[node]^2 * in[i]   (self-loop term; also initializes the buffer)
__global__ void k_selfloop(const float* __restrict__ hin, const float* __restrict__ dinv,
                           float* __restrict__ hout, int nfeat, long long total) {
  long long i = (long long)blockIdx.x * blockDim.x + threadIdx.x;
  if (i >= total) return;
  int node = (int)(i / nfeat);
  float d = dinv[node];
  hout[i] = d * d * hin[i];
}

// per-edge gather/scale/scatter-add, 4 features per thread (float4)
__global__ void k_edge_scatter(const float* __restrict__ hin, float* __restrict__ hout,
                               const int* __restrict__ src, const int* __restrict__ dst,
                               const float* __restrict__ dinv, int nfeat, int nchunk, int E) {
  long long gid = (long long)blockIdx.x * blockDim.x + threadIdx.x;
  int e = (int)(gid / nchunk);
  if (e >= E) return;
  int c = (int)(gid % nchunk);
  int s = src[e], d = dst[e];
  float coeff = dinv[s] * dinv[d];
  const float4 v = *(const float4*)(hin + (long long)s * nfeat + (long long)c * 4);
  float* o = hout + (long long)d * nfeat + (long long)c * 4;
  atomicAdd(o + 0, coeff * v.x);
  atomicAdd(o + 1, coeff * v.y);
  atomicAdd(o + 2, coeff * v.z);
  atomicAdd(o + 3, coeff * v.w);
}

__global__ void k_f32_to_bf16(const float* __restrict__ in, unsigned short* __restrict__ out,
                              long long n) {
  long long i = (long long)blockIdx.x * blockDim.x + threadIdx.x;
  if (i < n) out[i] = f2bf(in[i]);
}

// wt[n*K + k] = bf16(w[k*Ncol + n])   (transpose to column-major for WMMA B)
__global__ void k_transpose_bf16(const float* __restrict__ w, unsigned short* __restrict__ wt,
                                 int K, int Ncol) {
  int i = blockIdx.x * blockDim.x + threadIdx.x;
  if (i >= K * Ncol) return;
  int k = i / Ncol, n = i % Ncol;
  wt[(long long)n * K + k] = f2bf(w[i]);
}

// ---------------------------------------------------------------- WMMA GEMMs
// C[16r x 64] = A[16r x 128] @ W^T + bias ; one wave per 16x16 tile, 4 col tiles / block
__global__ __launch_bounds__(128) void k_gemm1_wmma(const unsigned short* __restrict__ A,
                                                    const unsigned short* __restrict__ Wt,
                                                    const float* __restrict__ bias,
                                                    float* __restrict__ C, int nRowTiles) {
  const int K = 128, NCOL = 64;
  int rt = blockIdx.x;
  if (rt >= nRowTiles) return;
  int wave = threadIdx.x >> 5;              // column tile 0..3
  int lane = threadIdx.x & 31;
  int m  = lane & 15;
  int kh = lane >> 4;
  const unsigned short* arow = A  + (long long)(rt * 16 + m) * K;
  const unsigned short* brow = Wt + (long long)(wave * 16 + m) * K;
  v8f acc = {};
#pragma unroll
  for (int k0 = 0; k0 < K; k0 += 32) {
    // A fragment: lane holds K = 8*kh..8*kh+7 and 16+8*kh..16+8*kh+7 of row m
    v8bf alo = *(const v8bf*)(arow + k0 + 8 * kh);
    v8bf ahi = *(const v8bf*)(arow + k0 + 16 + 8 * kh);
    v16bf a;
#pragma unroll
    for (int i = 0; i < 8; ++i) { a[i] = alo[i]; a[8 + i] = ahi[i]; }
    // B fragment: lane holds K = 8*kh..8*kh+7 and 16+8*kh..16+8*kh+7 of column m (Wt is col-major)
    v8bf blo = *(const v8bf*)(brow + k0 + 8 * kh);
    v8bf bhi = *(const v8bf*)(brow + k0 + 16 + 8 * kh);
    v16bf b;
#pragma unroll
    for (int i = 0; i < 8; ++i) { b[i] = blo[i]; b[8 + i] = bhi[i]; }
    acc = __builtin_amdgcn_wmma_f32_16x16x32_bf16(false, a, false, b,
                                                  (short)0, acc, false, false);
  }
  int n  = wave * 16 + m;
  float bn = bias[n];
  int rbase = 8 * kh;
#pragma unroll
  for (int r = 0; r < 8; ++r)
    C[(long long)(rt * 16 + rbase + r) * NCOL + n] = acc[r] + bn;
}

// C[16r x 16] = A[16r x 64] @ W^T ; one wave per row tile (no bias: added later)
__global__ __launch_bounds__(128) void k_gemm2_wmma(const unsigned short* __restrict__ A,
                                                    const unsigned short* __restrict__ Wt,
                                                    float* __restrict__ C, int nRowTiles) {
  const int K = 64, NCOL = 16;
  int rt = blockIdx.x * (blockDim.x >> 5) + (threadIdx.x >> 5);
  if (rt >= nRowTiles) return;                // wave-uniform
  int lane = threadIdx.x & 31;
  int m  = lane & 15;
  int kh = lane >> 4;
  const unsigned short* arow = A  + (long long)(rt * 16 + m) * K;
  const unsigned short* brow = Wt + (long long)m * K;
  v8f acc = {};
#pragma unroll
  for (int k0 = 0; k0 < K; k0 += 32) {
    v8bf alo = *(const v8bf*)(arow + k0 + 8 * kh);
    v8bf ahi = *(const v8bf*)(arow + k0 + 16 + 8 * kh);
    v16bf a;
#pragma unroll
    for (int i = 0; i < 8; ++i) { a[i] = alo[i]; a[8 + i] = ahi[i]; }
    v8bf blo = *(const v8bf*)(brow + k0 + 8 * kh);
    v8bf bhi = *(const v8bf*)(brow + k0 + 16 + 8 * kh);
    v16bf b;
#pragma unroll
    for (int i = 0; i < 8; ++i) { b[i] = blo[i]; b[8 + i] = bhi[i]; }
    acc = __builtin_amdgcn_wmma_f32_16x16x32_bf16(false, a, false, b,
                                                  (short)0, acc, false, false);
  }
  int rbase = 8 * kh;
#pragma unroll
  for (int r = 0; r < 8; ++r)
    C[(long long)(rt * 16 + rbase + r) * NCOL + m] = acc[r];
}

// ---------------------------------------------------------------- attention
// one wave per node; lane j handles attention hidden unit j (ATT_H == 32 == wave32)
__global__ void k_attention(float* __restrict__ h, const float* __restrict__ local,
                            const float* __restrict__ gtopo, const float* __restrict__ w1,
                            const float* __restrict__ ab1, const float* __restrict__ w2,
                            int n) {
  int wid  = (int)(((long long)blockIdx.x * blockDim.x + threadIdx.x) >> 5);
  int lane = threadIdx.x & 31;
  if (wid >= n) return;                       // wave-uniform
  const float* hr = h     + (long long)wid * 64;
  const float* lr = local + (long long)wid * 64;
  float s0 = ab1[lane], s1 = ab1[lane];
#pragma unroll 8
  for (int k = 0; k < 64; ++k) {
    float w = w1[k * 32 + lane];
    s0 = fmaf(hr[k], w, s0);
    s1 = fmaf(lr[k], w, s1);
  }
  float wk = w2[lane];
  float t0 = tanhf(s0) * wk;
  float t1 = tanhf(s1) * wk;
#pragma unroll
  for (int off = 16; off; off >>= 1) {
    t0 += __shfl_xor(t0, off, 32);
    t1 += __shfl_xor(t1, off, 32);
  }
  float mx = fmaxf(t0, t1);
  float e0 = __expf(t0 - mx), e1 = __expf(t1 - mx);
  float inv = 1.f / (e0 + e1);
  float b0 = e0 * inv, b1 = e1 * inv;
  float* ho = h + (long long)wid * 64;
#pragma unroll
  for (int f = lane; f < 64; f += 32)
    ho[f] = (b0 * hr[f] + b1 * lr[f]) * gtopo[f];
}

// ---------------------------------------------------------------- head
__global__ void k_logsoftmax16(const float* __restrict__ in, const float* __restrict__ bias,
                               float* __restrict__ out, int n) {
  int i = blockIdx.x * blockDim.x + threadIdx.x;
  if (i >= n) return;
  float v[16];
  float mx = -3.4e38f;
#pragma unroll
  for (int c = 0; c < 16; ++c) {
    v[c] = in[(long long)i * 16 + c] + bias[c];
    mx = fmaxf(mx, v[c]);
  }
  float s = 0.f;
#pragma unroll
  for (int c = 0; c < 16; ++c) s += __expf(v[c] - mx);
  float ls = __logf(s) + mx;
#pragma unroll
  for (int c = 0; c < 16; ++c) out[(long long)i * 16 + c] = v[c] - ls;
}

// ---------------------------------------------------------------- launch
static inline int cdiv_ll(long long a, long long b) { return (int)((a + b - 1) / b); }

extern "C" void kernel_launch(void* const* d_in, const int* in_sizes, int n_in,
                              void* d_out, int out_size, void* d_ws, size_t ws_size,
                              hipStream_t stream) {
  const float* x       = (const float*)d_in[0];
  const int*   ei      = (const int*)  d_in[1];
  const float* local_t = (const float*)d_in[2];
  const float* global_t= (const float*)d_in[3];
  const float* w_conv1 = (const float*)d_in[4];
  const float* b_conv1 = (const float*)d_in[5];
  const float* w_gc2   = (const float*)d_in[6];
  const float* b_gc2   = (const float*)d_in[7];
  const float* att_w1  = (const float*)d_in[8];
  const float* att_b1  = (const float*)d_in[9];
  const float* att_w2  = (const float*)d_in[10];

  const int N = in_sizes[0] / 128;     // 50000
  const int E = in_sizes[1] / 2;       // 1600000
  const int* src = ei;
  const int* dst = ei + E;

  // ---- workspace carve-up (bytes, 256-aligned)
  char* ws = (char*)d_ws;
  size_t off = 0;
  auto take = [&](size_t bytes) { char* p = ws + off; off = (off + bytes + 255) & ~(size_t)255; return p; };
  float* dinv = (float*)take((size_t)N * 4);
  char*  R1   = take((size_t)N * 128 * 4);
  char*  R2   = take((size_t)N * 128 * 4);
  unsigned short* Wt1 = (unsigned short*)take((size_t)64 * 128 * 2);
  unsigned short* Wt2 = (unsigned short*)take((size_t)16 * 64 * 2);
  (void)ws_size; (void)n_in; (void)out_size;

  float* tmp1 = (float*)R1;                                   // hop-1 result (N x 128)
  float* tmp2 = (float*)R2;                                   // hop-2 result (N x 128)
  unsigned short* Abf   = (unsigned short*)R1;                // bf16(tmp2): N x 128
  float*          h_hid = (float*)R2;                         // conv1 / attention: N x 64
  unsigned short* Hbf   = (unsigned short*)(R1 + 12800000);   // bf16(h_att): N x 64
  float* cls_pre = (float*)(R1 + 19200000);                   // N x 16
  float* cls_agg = (float*)(R1 + 22400000);                   // N x 16
  float* out = (float*)d_out;

  const int T = 256;
  // 1) degree / dinv
  k_zero_f32 <<<cdiv_ll(N, T), T, 0, stream>>>(dinv, N);
  k_deg_count<<<cdiv_ll(E, T), T, 0, stream>>>(dst, dinv, E);
  k_dinv     <<<cdiv_ll(N, T), T, 0, stream>>>(dinv, N);

  // 2) hop 1: tmp1 = P x
  k_selfloop    <<<cdiv_ll((long long)N * 128, T), T, 0, stream>>>(x, dinv, tmp1, 128, (long long)N * 128);
  k_edge_scatter<<<cdiv_ll((long long)E * 32,  T), T, 0, stream>>>(x, tmp1, src, dst, dinv, 128, 32, E);
  // 3) hop 2: tmp2 = P tmp1
  k_selfloop    <<<cdiv_ll((long long)N * 128, T), T, 0, stream>>>(tmp1, dinv, tmp2, 128, (long long)N * 128);
  k_edge_scatter<<<cdiv_ll((long long)E * 32,  T), T, 0, stream>>>(tmp1, tmp2, src, dst, dinv, 128, 32, E);

  // 4) bf16 conversions for WMMA
  k_f32_to_bf16   <<<cdiv_ll((long long)N * 128, T), T, 0, stream>>>(tmp2, Abf, (long long)N * 128);
  k_transpose_bf16<<<cdiv_ll(128 * 64, T), T, 0, stream>>>(w_conv1, Wt1, 128, 64);
  k_transpose_bf16<<<cdiv_ll(64 * 16,  T), T, 0, stream>>>(w_gc2,   Wt2, 64, 16);

  // 5) h_hid = tmp2 @ w_conv1 + b   (WMMA bf16, 3125 row tiles x 4 col tiles)
  k_gemm1_wmma<<<N / 16, 128, 0, stream>>>(Abf, Wt1, b_conv1, h_hid, N / 16);

  // 6) attention aggregation (in place on h_hid), then bf16
  k_attention  <<<cdiv_ll((long long)N * 32, T), T, 0, stream>>>(h_hid, local_t, global_t,
                                                                 att_w1, att_b1, att_w2, N);
  k_f32_to_bf16<<<cdiv_ll((long long)N * 64, T), T, 0, stream>>>(h_hid, Hbf, (long long)N * 64);

  // 7) cls_pre = h_att @ w_gc2 (WMMA), then one 16-wide propagation hop
  k_gemm2_wmma  <<<cdiv_ll(N / 16, 4), 128, 0, stream>>>(Hbf, Wt2, cls_pre, N / 16);
  k_selfloop    <<<cdiv_ll((long long)N * 16, T), T, 0, stream>>>(cls_pre, dinv, cls_agg, 16, (long long)N * 16);
  k_edge_scatter<<<cdiv_ll((long long)E * 4,  T), T, 0, stream>>>(cls_pre, cls_agg, src, dst, dinv, 16, 4, E);

  // 8) + bias, log_softmax -> out
  k_logsoftmax16<<<cdiv_ll(N, T), T, 0, stream>>>(cls_agg, b_gc2, out, N);
}
